// GPSA_1039382086292
// MI455X (gfx1250) — matrix operand order repaired and checked by reading, not compiled, not run on hardware
//
#include <hip/hip_runtime.h>
#include <hip/hip_bf16.h>
#include <stdint.h>

typedef __attribute__((ext_vector_type(16))) __bf16 v16bf;
typedef __attribute__((ext_vector_type(8)))  float  v8f;

#define BATCH 8
#define SEQ   1024
#define DMODEL 384
#define NH    6
#define DH    64
#define BN    (BATCH * SEQ)
#define XS    40   // LDS row stride (bf16) for 32-wide K tiles: 80B = 5*16B
#define KS    72   // LDS row stride (bf16) for 64-wide tiles:  144B = 9*16B

union Frag {
    v16bf v;
    uint4 q[2];
};

__device__ __forceinline__ uint16_t f2bf(float f) {
    uint32_t b = __float_as_uint(f);
    b += 0x7FFFu + ((b >> 16) & 1u);       // round-to-nearest-even
    return (uint16_t)(b >> 16);
}

__device__ __forceinline__ v8f zero8() {
    v8f z = {0.f, 0.f, 0.f, 0.f, 0.f, 0.f, 0.f, 0.f};
    return z;
}

__device__ __forceinline__ v8f wmma_bf16(const Frag& a, const Frag& b, v8f c) {
    return __builtin_amdgcn_wmma_f32_16x16x32_bf16(false, a.v, false, b.v,
                                                   (short)0, c, false, false);
}

// Async 16B copy global -> LDS (CDNA5 TDM-adjacent path, ASYNCcnt-tracked).
// Low 32 bits of a generic pointer to __shared__ == LDS byte offset.
__device__ __forceinline__ void async_ld_b128(uint16_t* lds_dst, const uint16_t* gsrc) {
    uint32_t ldsoff = (uint32_t)(uintptr_t)lds_dst;
    asm volatile("global_load_async_to_lds_b128 %0, %1, off"
                 :: "v"(ldsoff), "v"(gsrc) : "memory");
}
__device__ __forceinline__ void wait_async_0() {
    asm volatile("s_wait_asynccnt 0x0" ::: "memory");
}
__device__ __forceinline__ void wait_async_4() {
    asm volatile("s_wait_asynccnt 0x4" ::: "memory");
}

// ---------------------------------------------------------------------------
// Kernel 1: fused QKV projection.  y = x @ W.T  (W row-major (Dout,Din))
// grid (BN/128, NH, 3), block 256.
// Q,K written bf16 (B,h,N,dh); V written TRANSPOSED bf16 (B,h,dh,N) so the
// attention kernel can stage both K and V tiles as plain async row copies.
// ---------------------------------------------------------------------------
__global__ __launch_bounds__(256)
void qkv_proj_kernel(const float* __restrict__ x,
                     const float* __restrict__ Wq,
                     const float* __restrict__ Wk,
                     const float* __restrict__ Wv,
                     uint16_t* __restrict__ qkv_ws) {
    __shared__ __align__(16) uint16_t Abuf[128 * XS];
    __shared__ __align__(16) uint16_t Bbuf[64 * XS];
    __shared__ __align__(16) uint16_t Tbuf[128 * KS];   // transpose staging (V)

    const int tid = threadIdx.x;
    const int wave = tid >> 5, lane = tid & 31, l15 = lane & 15, hi = lane >> 4;
    const int rb = blockIdx.x * 128;
    const int h  = blockIdx.y;
    const int z  = blockIdx.z;
    const float* W = (z == 0) ? Wq : (z == 1) ? Wk : Wv;
    uint16_t* outp = qkv_ws + (size_t)z * (BATCH * NH * SEQ * DH);

    v8f acc[4];
#pragma unroll
    for (int t = 0; t < 4; ++t) acc[t] = zero8();

    for (int kb = 0; kb < DMODEL; kb += 32) {
        __syncthreads();
#pragma unroll
        for (int p = 0; p < 4; ++p) {           // x tile: 128x32 f32 -> bf16
            int idx = tid + p * 256;
            int row = idx >> 3;
            int c4  = (idx & 7) * 4;
            float4 xv = *(const float4*)(x + (size_t)(rb + row) * DMODEL + kb + c4);
            uint16_t* dst = &Abuf[row * XS + c4];
            dst[0] = f2bf(xv.x); dst[1] = f2bf(xv.y);
            dst[2] = f2bf(xv.z); dst[3] = f2bf(xv.w);
        }
#pragma unroll
        for (int p = 0; p < 2; ++p) {           // W tile: 64x32 f32 -> bf16
            int idx = tid + p * 256;
            int row = idx >> 3;
            int c4  = (idx & 7) * 4;
            float4 wv = *(const float4*)(W + (size_t)(h * 64 + row) * DMODEL + kb + c4);
            uint16_t* dst = &Bbuf[row * XS + c4];
            dst[0] = f2bf(wv.x); dst[1] = f2bf(wv.y);
            dst[2] = f2bf(wv.z); dst[3] = f2bf(wv.w);
        }
        __syncthreads();

        Frag a;                                  // A 16x32: lanes 0-15 K{0-7,16-23}
        const uint16_t* ap = &Abuf[(wave * 16 + l15) * XS + 8 * hi];
        a.q[0] = *(const uint4*)(ap);
        a.q[1] = *(const uint4*)(ap + 16);
#pragma unroll
        for (int t = 0; t < 4; ++t) {            // B 32x16: lane=N, lanes 0-15 K{0-7,16-23}
            Frag b;
            const uint16_t* bp = &Bbuf[(t * 16 + l15) * XS + 8 * hi];
            b.q[0] = *(const uint4*)(bp);
            b.q[1] = *(const uint4*)(bp + 16);
            acc[t] = wmma_bf16(a, b, acc[t]);
        }
    }

    if (z == 2) {
        // V: transpose 128x64 tile through LDS, store (B,h,dh,N) with b128s.
#pragma unroll
        for (int t = 0; t < 4; ++t)
#pragma unroll
            for (int r = 0; r < 8; ++r)
                Tbuf[(wave * 16 + r + 8 * hi) * KS + t * 16 + l15] = f2bf(acc[t][r]);
        __syncthreads();
        const int b = rb >> 10, n0blk = rb & 1023;
#pragma unroll
        for (int p = 0; p < 4; ++p) {
            int idx = tid + p * 256;            // 0..1023
            int d  = idx >> 4;                  // dh column 0..63
            int n0 = (idx & 15) * 8;            // n chunk
            uint32_t w0 = (uint32_t)Tbuf[(n0 + 0) * KS + d] |
                          ((uint32_t)Tbuf[(n0 + 1) * KS + d] << 16);
            uint32_t w1 = (uint32_t)Tbuf[(n0 + 2) * KS + d] |
                          ((uint32_t)Tbuf[(n0 + 3) * KS + d] << 16);
            uint32_t w2 = (uint32_t)Tbuf[(n0 + 4) * KS + d] |
                          ((uint32_t)Tbuf[(n0 + 5) * KS + d] << 16);
            uint32_t w3 = (uint32_t)Tbuf[(n0 + 6) * KS + d] |
                          ((uint32_t)Tbuf[(n0 + 7) * KS + d] << 16);
            uint4 val; val.x = w0; val.y = w1; val.z = w2; val.w = w3;
            *(uint4*)(outp + ((size_t)(b * NH + h) * DH + d) * SEQ + n0blk + n0) = val;
        }
    } else {
        const int rowb = rb + wave * 16;
#pragma unroll
        for (int t = 0; t < 4; ++t)
#pragma unroll
            for (int r = 0; r < 8; ++r) {
                int row = rowb + r + 8 * hi;
                int bb = row >> 10, n = row & 1023;
                outp[((size_t)(bb * NH + h) * SEQ + n) * DH + t * 16 + l15] =
                    f2bf(acc[t][r]);
            }
    }
}

// ---------------------------------------------------------------------------
// Kernel 2: flash attention with Gaussian positional blend.
// grid (SEQ/128, BATCH*NH), block 256 (8 waves, 16 q-rows per wave).
// K (B,h,N,dh) and V (B,h,dh,N) tiles staged via double-buffered async
// global->LDS copies (ASYNCcnt) so HBM/L2 latency hides behind WMMAs.
// ---------------------------------------------------------------------------
__global__ __launch_bounds__(256)
void attn_kernel(const uint16_t* __restrict__ Qw,
                 const uint16_t* __restrict__ Kw,
                 const uint16_t* __restrict__ Vw,
                 const float* __restrict__ logit_lambda,
                 const float* __restrict__ log_sigma,
                 uint16_t* __restrict__ AO) {
    __shared__ __align__(16) uint16_t KbufD[2][64 * KS];
    __shared__ __align__(16) uint16_t VbufD[2][64 * KS];  // [dh][kv]
    __shared__ __align__(16) uint16_t Pbuf[8][16 * KS];   // per-wave prob tiles

    const int tid = threadIdx.x;
    const int wave = tid >> 5, lane = tid & 31, l15 = lane & 15, hi = lane >> 4;
    const int qb = blockIdx.x * 128;
    const int bh = blockIdx.y;
    const size_t base = (size_t)bh * SEQ * DH;   // same flat offset for K and V

    const int crow = (tid >> 3);                 // staging row 0..31 (x2 passes)
    const int cq   = (tid & 7) * 8;              // staging 16B chunk

    const float lam   = 1.0f / (1.0f + __expf(-logit_lambda[0]));
    const float sigma = __logf(1.0f + __expf(log_sigma[0])) + 1e-6f;
    const float coef  = 1.0f / (2.0f * sigma * sigma);
    const float invsqdh = 0.125f;                // 1/sqrt(64)

    Frag qa[2];                                  // Q 16x64 held in registers
    const int qrow = qb + wave * 16 + l15;
    const uint16_t* qp = Qw + base + (size_t)qrow * DH;
#pragma unroll
    for (int c = 0; c < 2; ++c) {
        qa[c].q[0] = *(const uint4*)(qp + 32 * c + 8 * hi);
        qa[c].q[1] = *(const uint4*)(qp + 32 * c + 16 + 8 * hi);
    }

    v8f acc_o[4];
    float m[8], lsum[8];
#pragma unroll
    for (int t = 0; t < 4; ++t) acc_o[t] = zero8();
#pragma unroll
    for (int r = 0; r < 8; ++r) { m[r] = -1e30f; lsum[r] = 0.f; }

    const int myrow0 = qb + wave * 16 + 8 * hi;  // rows: myrow0 + r (r=0..7)

    // prologue: async-issue stage 0 (4 async instrs per wave: 2 K + 2 V)
#pragma unroll
    for (int p = 0; p < 2; ++p) {
        int row = crow + 32 * p;
        async_ld_b128(&KbufD[0][row * KS + cq], Kw + base + (size_t)row * DH + cq);
        async_ld_b128(&VbufD[0][row * KS + cq], Vw + base + (size_t)row * SEQ + cq);
    }

    for (int it = 0; it < SEQ / 64; ++it) {
        const int j0 = it * 64;
        const uint16_t* Kbuf = KbufD[it & 1];
        const uint16_t* Vbuf = VbufD[it & 1];

        if (it + 1 < SEQ / 64) {                 // prefetch next KV block
            const int jn = j0 + 64;
            uint16_t* kd = KbufD[(it + 1) & 1];
            uint16_t* vd = VbufD[(it + 1) & 1];
#pragma unroll
            for (int p = 0; p < 2; ++p) {
                int row = crow + 32 * p;
                async_ld_b128(&kd[row * KS + cq],
                              Kw + base + (size_t)(jn + row) * DH + cq);
                async_ld_b128(&vd[row * KS + cq],
                              Vw + base + (size_t)row * SEQ + jn + cq);
            }
            wait_async_4();                      // oldest 4 (= stage it) done
        } else {
            wait_async_0();
        }
        __syncthreads();                         // all waves' stage it visible

        v8f s[4];                                // S = Q K^T  (4 n-subtiles)
#pragma unroll
        for (int t = 0; t < 4; ++t) {
            v8f a0 = zero8();
#pragma unroll
            for (int c = 0; c < 2; ++c) {
                Frag kb;
                const uint16_t* kp = &Kbuf[(t * 16 + l15) * KS + 32 * c + 8 * hi];
                kb.q[0] = *(const uint4*)(kp);
                kb.q[1] = *(const uint4*)(kp + 16);
                a0 = wmma_bf16(qa[c], kb, a0);
            }
            s[t] = a0;
        }

        // logits = lam*content + (1-lam)*exp(-dist2/(2 sigma^2))
#pragma unroll
        for (int t = 0; t < 4; ++t) {
            int col = j0 + t * 16 + l15;
            float yj = (float)(col >> 5), xj = (float)(col & 31);
#pragma unroll
            for (int r = 0; r < 8; ++r) {
                int rowi = myrow0 + r;
                float dy = (float)(rowi >> 5) - yj;
                float dx = (float)(rowi & 31) - xj;
                float P = __expf(-(dy * dy + dx * dx) * coef);
                s[t][r] = lam * (s[t][r] * invsqdh) + (1.0f - lam) * P;
            }
        }

        // online softmax update (row = 16 lanes of one half-wave)
        float alpha[8];
#pragma unroll
        for (int r = 0; r < 8; ++r) {
            float mx = fmaxf(fmaxf(s[0][r], s[1][r]), fmaxf(s[2][r], s[3][r]));
#pragma unroll
            for (int d = 1; d < 16; d <<= 1) mx = fmaxf(mx, __shfl_xor(mx, d, 32));
            float mn = fmaxf(m[r], mx);
            alpha[r] = __expf(m[r] - mn);
            m[r] = mn;
            float rs = 0.f;
#pragma unroll
            for (int t = 0; t < 4; ++t) {
                float p = __expf(s[t][r] - mn);
                s[t][r] = p;
                rs += p;
            }
#pragma unroll
            for (int d = 1; d < 16; d <<= 1) rs += __shfl_xor(rs, d, 32);
            lsum[r] = lsum[r] * alpha[r] + rs;
        }
#pragma unroll
        for (int t = 0; t < 4; ++t)
#pragma unroll
            for (int r = 0; r < 8; ++r)
                acc_o[t][r] = acc_o[t][r] * alpha[r];

        // probs: C-layout -> per-wave LDS tile (row-major) for A-fragment reads
        uint16_t* pb = Pbuf[wave];
#pragma unroll
        for (int t = 0; t < 4; ++t)
#pragma unroll
            for (int r = 0; r < 8; ++r)
                pb[(r + 8 * hi) * KS + t * 16 + l15] = f2bf(s[t][r]);

        // O += P @ V
#pragma unroll
        for (int c = 0; c < 2; ++c) {
            Frag pa;
            const uint16_t* pp = &pb[l15 * KS + 32 * c + 8 * hi];
            pa.q[0] = *(const uint4*)(pp);
            pa.q[1] = *(const uint4*)(pp + 16);
#pragma unroll
            for (int t = 0; t < 4; ++t) {
                Frag vb;
                const uint16_t* vp = &Vbuf[(t * 16 + l15) * KS + 32 * c + 8 * hi];
                vb.q[0] = *(const uint4*)(vp);
                vb.q[1] = *(const uint4*)(vp + 16);
                acc_o[t] = wmma_bf16(pa, vb, acc_o[t]);
            }
        }
        __syncthreads();   // everyone done reading buf[it&1] before it's rewritten
    }

    // epilogue: O/l -> bf16 attn_out in (B, N, h*dh) layout
    const int b = bh / NH, h = bh % NH;
#pragma unroll
    for (int r = 0; r < 8; ++r) {
        float inv = 1.0f / lsum[r];
        int n = qb + wave * 16 + r + 8 * hi;
        uint16_t* op = AO + (size_t)(b * SEQ + n) * DMODEL + h * DH;
#pragma unroll
        for (int t = 0; t < 4; ++t)
            op[t * 16 + l15] = f2bf(acc_o[t][r] * inv);
    }
}

// ---------------------------------------------------------------------------
// Kernel 3: output projection.  out = attn_out @ Wp.T   (f32 output)
// grid (BN/128, DMODEL/64), block 256.  bf16 A tile staged via async copies.
// ---------------------------------------------------------------------------
__global__ __launch_bounds__(256)
void out_proj_kernel(const uint16_t* __restrict__ ao,
                     const float* __restrict__ Wp,
                     float* __restrict__ out) {
    __shared__ __align__(16) uint16_t Abuf[128 * XS];
    __shared__ __align__(16) uint16_t Bbuf[64 * XS];

    const int tid = threadIdx.x;
    const int wave = tid >> 5, lane = tid & 31, l15 = lane & 15, hi = lane >> 4;
    const int rb = blockIdx.x * 128;
    const int cb = blockIdx.y * 64;

    v8f acc[4];
#pragma unroll
    for (int t = 0; t < 4; ++t) acc[t] = zero8();

    for (int kb = 0; kb < DMODEL; kb += 32) {
        __syncthreads();
#pragma unroll
        for (int p = 0; p < 2; ++p) {            // attn_out tile: async -> LDS
            int idx = tid + p * 256;
            int row = idx >> 2;
            int c8  = (idx & 3) * 8;
            async_ld_b128(&Abuf[row * XS + c8],
                          ao + (size_t)(rb + row) * DMODEL + kb + c8);
        }
#pragma unroll
        for (int p = 0; p < 2; ++p) {            // Wp tile f32 -> bf16 (manual)
            int idx = tid + p * 256;
            int row = idx >> 3;
            int c4  = (idx & 7) * 4;
            float4 wv = *(const float4*)(Wp + (size_t)(cb + row) * DMODEL + kb + c4);
            uint16_t* dst = &Bbuf[row * XS + c4];
            dst[0] = f2bf(wv.x); dst[1] = f2bf(wv.y);
            dst[2] = f2bf(wv.z); dst[3] = f2bf(wv.w);
        }
        wait_async_0();
        __syncthreads();

        Frag a;
        const uint16_t* ap = &Abuf[(wave * 16 + l15) * XS + 8 * hi];
        a.q[0] = *(const uint4*)(ap);
        a.q[1] = *(const uint4*)(ap + 16);
#pragma unroll
        for (int t = 0; t < 4; ++t) {
            Frag b;
            const uint16_t* bp = &Bbuf[(t * 16 + l15) * XS + 8 * hi];
            b.q[0] = *(const uint4*)(bp);
            b.q[1] = *(const uint4*)(bp + 16);
            acc[t] = wmma_bf16(a, b, acc[t]);
        }
    }

    const int rowb = rb + wave * 16;
#pragma unroll
    for (int t = 0; t < 4; ++t)
#pragma unroll
        for (int r = 0; r < 8; ++r) {
            int row = rowb + r + 8 * hi;
            out[(size_t)row * DMODEL + cb + t * 16 + l15] = acc[t][r];
        }
}

// ---------------------------------------------------------------------------
extern "C" void kernel_launch(void* const* d_in, const int* in_sizes, int n_in,
                              void* d_out, int out_size, void* d_ws, size_t ws_size,
                              hipStream_t stream) {
    const float* x  = (const float*)d_in[0];
    const float* Wq = (const float*)d_in[1];
    const float* Wk = (const float*)d_in[2];
    const float* Wv = (const float*)d_in[3];
    const float* Wp = (const float*)d_in[4];
    const float* ll = (const float*)d_in[5];
    const float* ls = (const float*)d_in[6];
    float* out = (float*)d_out;

    const size_t HEAD_ELEMS = (size_t)BATCH * NH * SEQ * DH;   // 3,145,728
    uint16_t* Qw = (uint16_t*)d_ws;
    uint16_t* Kw = Qw + HEAD_ELEMS;
    uint16_t* Vw = Kw + HEAD_ELEMS;   // transposed (B,h,dh,N)
    uint16_t* AO = Vw + HEAD_ELEMS;

    qkv_proj_kernel<<<dim3(BN / 128, NH, 3), 256, 0, stream>>>(x, Wq, Wk, Wv, Qw);
    attn_kernel<<<dim3(SEQ / 128, BATCH * NH), 256, 0, stream>>>(Qw, Kw, Vw, ll, ls, AO);
    out_proj_kernel<<<dim3(BN / 128, DMODEL / 64), 256, 0, stream>>>(AO, Wp, out);
}
